// Propagation_15582141350262
// MI455X (gfx1250) — matrix operand order repaired and checked by reading, not compiled, not run on hardware
//
#include <hip/hip_runtime.h>
#include <stdint.h>

typedef _Float16 half_t;
typedef __attribute__((ext_vector_type(16))) _Float16 v16h;
typedef __attribute__((ext_vector_type(8)))  float    v8f;
typedef __attribute__((ext_vector_type(4)))  uint32_t v4u;
typedef __attribute__((ext_vector_type(8)))  int      v8i;
typedef __attribute__((ext_vector_type(4)))  int      v4i;

#define CCH 256      // channels
#define NPIX 4096    // h*w
#define BT 64        // n/m tile width
#define STRX 264     // halves stride for [pixel][c] tiles (= 128 dw + 4 dw pad)
#define STRV 72      // halves stride for [c][m] value tile & [n][m] P tile (= 32 dw + 4 dw pad)

#if __has_builtin(__builtin_amdgcn_tensor_load_to_lds) && \
    __has_builtin(__builtin_amdgcn_s_wait_tensorcnt)
#define USE_TDM 1
#else
#define USE_TDM 0
#endif

// ---------------- WMMA helpers ----------------

__device__ __forceinline__ v8f wmma_f16(v16h a, v16h b, v8f c) {
  return __builtin_amdgcn_wmma_f32_16x16x32_f16(
      false, a, false, b, (short)0, c, false, false);
}

// A fragment (16x32 f16), LDS row-major (row=M, contiguous K). Two ds_load_b128.
__device__ __forceinline__ v16h frag_a(const half_t* base, int row0, int stride,
                                       int k0, int r, int hi) {
  const uint4* p = (const uint4*)(base + (row0 + r) * stride + k0 + hi * 8);
  union { v16h v; uint4 q[2]; } f;
  f.q[0] = p[0];   // halves k0 + hi*8 + 0..7
  f.q[1] = p[2];   // halves k0 + 16 + hi*8 + 0..7
  return f.v;
}

// B fragment (32x16 f16), LDS stored transposed (row=N col, contiguous K).
__device__ __forceinline__ v16h frag_b(const half_t* base, int col0, int stride,
                                       int k0, int r, int hi) {
  const uint4* p = (const uint4*)(base + (col0 + r) * stride + k0 + hi * 8);
  union { v16h v; uint4 q[2]; } f;
  f.q[0] = p[0];
  f.q[1] = p[2];
  return f.v;
}

union h8 { uint4 q; half_t h[8]; };

#if USE_TDM
// Async 2D tile DMA global->LDS via Tensor Data Mover (D# per CDNA5 ISA ch.8).
// Tile: tile_d1 rows x tile_d0 elements (2B each), global row stride = stride0
// elements, LDS padded by (pa_code+1) dwords after every 2^(pi_code+1) dwords.
__device__ __forceinline__ void tdm_load_2d(uint32_t lds_addr, const void* gaddr,
                                            uint32_t td0, uint32_t td1,
                                            uint32_t stride0, uint32_t tile_d0,
                                            uint32_t tile_d1, uint32_t pi_code,
                                            uint32_t pa_code) {
  const uint64_t ga = (uint64_t)(uintptr_t)gaddr;
  v4u g0 = {1u,                                    // count=1, user descriptor
            lds_addr,                              // LDS byte address
            (uint32_t)ga,                          // global_addr[31:0]
            (uint32_t)((ga >> 32) & 0x1FFFFFFull) | (2u << 30)};  // type=2
  v8i g1 = {(int)((1u << 16) | (1u << 20) | (pi_code << 22) | (pa_code << 25)),
            // ^ data_size=2B, pad_enable, pad_interval, pad_amount
            (int)((td0 & 0xFFFFu) << 16),                       // tensor_dim0 lo
            (int)((td0 >> 16) | ((td1 & 0xFFFFu) << 16)),       // dim0 hi, dim1 lo
            (int)((td1 >> 16) | (tile_d0 << 16)),               // dim1 hi, tile_dim0
            (int)(tile_d1 & 0xFFFFu),                           // tile_dim1 (tile_dim2=0)
            (int)stride0,                                       // tensor_dim0_stride lo
            0, 0};
  v4i z4 = {0, 0, 0, 0};
#if __clang_major__ >= 23
  v8i z8 = {0, 0, 0, 0, 0, 0, 0, 0};
  __builtin_amdgcn_tensor_load_to_lds(g0, g1, z4, z4, z8, 0);
#else
  __builtin_amdgcn_tensor_load_to_lds(g0, g1, z4, z4, 0);
#endif
}
#endif

// ---------------- Kernel 1: xnT[b][n][c] = l2norm_c(x) -> f16, pixel-major ----------------

__global__ __launch_bounds__(256) void norm_x_kernel(const float* __restrict__ x,
                                                     half_t* __restrict__ xnT) {
  const int gid = blockIdx.x * 256 + threadIdx.x;  // 4*4096 pixels
  const int b = gid >> 12;
  const int n = gid & (NPIX - 1);
  const float* xb = x + (size_t)b * CCH * NPIX + n;
  float s = 0.f;
#pragma unroll 4
  for (int c = 0; c < CCH; ++c) { float v = xb[(size_t)c * NPIX]; s = fmaf(v, v, s); }
  const float sc = 1.f / fmaxf(sqrtf(s), 1e-12f);
  half_t* ot = xnT + ((size_t)b * NPIX + n) * CCH;   // transposed: contiguous channels
#pragma unroll 4
  for (int cv = 0; cv < CCH; cv += 8) {
    h8 pk;
#pragma unroll
    for (int j = 0; j < 8; ++j) pk.h[j] = (half_t)(xb[(size_t)(cv + j) * NPIX] * sc);
    *(uint4*)&ot[cv] = pk.q;
  }
}

// ---------------- Kernel 2: value = l2norm_c(W@x + bias) -> f16 (channel-major) ----------------

__global__ __launch_bounds__(256) void conv_value_kernel(const float* __restrict__ x,
                                                         const float* __restrict__ W,
                                                         const float* __restrict__ bias,
                                                         half_t* __restrict__ val) {
  __shared__ float xs[CCH][68];
  __shared__ float partial[4][64];
  __shared__ float scl[64];
  const int b = blockIdx.y;
  const int n0 = blockIdx.x * 64;
  const int tid = threadIdx.x;
  const float* xb = x + (size_t)b * CCH * NPIX;

  for (int idx = tid; idx < CCH * 16; idx += 256) {
    const int c = idx >> 4, p4 = idx & 15;
    *(float4*)&xs[c][p4 * 4] = *(const float4*)&xb[(size_t)c * NPIX + n0 + p4 * 4];
  }
  __syncthreads();

  const int o = tid;
  float acc[64];
  const float bo = bias[o];
#pragma unroll
  for (int p = 0; p < 64; ++p) acc[p] = bo;
  const float* wr = W + (size_t)o * CCH;
  for (int i = 0; i < CCH; ++i) {
    const float w = wr[i];
    const float4* row = (const float4*)&xs[i][0];
#pragma unroll
    for (int pv = 0; pv < 16; ++pv) {
      float4 v = row[pv];
      acc[pv * 4 + 0] = fmaf(w, v.x, acc[pv * 4 + 0]);
      acc[pv * 4 + 1] = fmaf(w, v.y, acc[pv * 4 + 1]);
      acc[pv * 4 + 2] = fmaf(w, v.z, acc[pv * 4 + 2]);
      acc[pv * 4 + 3] = fmaf(w, v.w, acc[pv * 4 + 3]);
    }
  }
  __syncthreads();
#pragma unroll
  for (int p = 0; p < 64; ++p) xs[o][p] = acc[p];
  __syncthreads();

  const int q = tid >> 6, pp = tid & 63;
  float s = 0.f;
  for (int oo = q * 64; oo < q * 64 + 64; ++oo) { float v = xs[oo][pp]; s = fmaf(v, v, s); }
  partial[q][pp] = s;
  __syncthreads();
  if (tid < 64) {
    float t = partial[0][tid] + partial[1][tid] + partial[2][tid] + partial[3][tid];
    scl[tid] = 1.f / fmaxf(sqrtf(t), 1e-12f);
  }
  __syncthreads();

  half_t* vb = val + (size_t)b * CCH * NPIX + (size_t)o * NPIX + n0;
#pragma unroll
  for (int pv = 0; pv < 8; ++pv) {
    h8 pk;
#pragma unroll
    for (int j = 0; j < 8; ++j) pk.h[j] = (half_t)(acc[pv * 8 + j] * scl[pv * 8 + j]);
    *(uint4*)&vb[pv * 8] = pk.q;
  }
}

// ---------------- Kernel 3: fused  out = value @ (relu(xn^T xn)^2)^T ----------------
// Double-buffered m-tiles; TDM prefetches tile m+64 while WMMA consumes tile m.

__global__ __launch_bounds__(256) void prop_attn_kernel(const half_t* __restrict__ xnT,
                                                        const half_t* __restrict__ val,
                                                        float* __restrict__ out) {
  __shared__ half_t sX[2][BT * STRX];   // [pixel][c] tiles; slot0 first holds the n-tile
  __shared__ half_t sVb[2][CCH * STRV]; // [c][m] value tiles
  __shared__ half_t sP[BT * STRV];      // [n][m] relu^2 scores

  const int b = blockIdx.y;
  const int n0 = blockIdx.x * BT;
  const int tid = threadIdx.x;
  const int lane = tid & 31;
  const int wave = tid >> 5;
  const int r = lane & 15;
  const int hi = lane >> 4;

  const half_t* xt = xnT + (size_t)b * NPIX * CCH;  // pixel-major
  const half_t* vb = val + (size_t)b * CCH * NPIX;  // channel-major

#if USE_TDM
  if (wave == 0) {
    // n-tile of xn^T into slot 0; first m-tile (+its value tile) into slot 1.
    tdm_load_2d((uint32_t)(uintptr_t)&sX[0][0], xt + (size_t)n0 * CCH,
                CCH, NPIX, CCH, CCH, BT, /*pi=128dw*/ 6, /*pa=4dw*/ 3);
    tdm_load_2d((uint32_t)(uintptr_t)&sX[1][0], xt,
                CCH, NPIX, CCH, CCH, BT, 6, 3);
    tdm_load_2d((uint32_t)(uintptr_t)&sVb[1][0], vb,
                NPIX, CCH, NPIX, BT, CCH, /*pi=32dw*/ 4, /*pa=4dw*/ 3);
    __builtin_amdgcn_s_wait_tensorcnt(0);
  }
  __syncthreads();
#else
  for (int idx = tid; idx < (CCH * BT) / 8; idx += 256) {
    const int row = idx >> 5, cv = (idx & 31) * 8;       // sX copy shape
    *(uint4*)&sX[0][row * STRX + cv] = *(const uint4*)&xt[(size_t)(n0 + row) * CCH + cv];
    *(uint4*)&sX[1][row * STRX + cv] = *(const uint4*)&xt[(size_t)row * CCH + cv];
    const int c = idx >> 3, mv = (idx & 7) * 8;          // sV copy shape
    *(uint4*)&sVb[1][c * STRV + mv] = *(const uint4*)&vb[(size_t)c * NPIX + mv];
  }
  __syncthreads();
#endif

  // mm1 tile assignment: 16 S tiles (4x4), 2 per wave sharing the same n row.
  const int ni = wave >> 1;
  const int mi0 = (wave & 1) * 2;
  const int c0 = wave * 32;  // mm2: each wave owns 32 output channels

  // Resident mm1 A fragments (the n-tile is reused by all 64 m-steps).
  v16h Amm1[8];
#pragma unroll
  for (int ks = 0; ks < 8; ++ks) Amm1[ks] = frag_a(sX[0], ni * 16, STRX, ks * 32, r, hi);
  __syncthreads();  // slot 0 free for reuse after this

  const v8f vzero = {0.f, 0.f, 0.f, 0.f, 0.f, 0.f, 0.f, 0.f};
  v8f Oacc[2][4];
#pragma unroll
  for (int i = 0; i < 2; ++i)
#pragma unroll
    for (int j = 0; j < 4; ++j) Oacc[i][j] = vzero;

  int cur = 1;
  for (int m0 = 0; m0 < NPIX; m0 += BT) {
    const int nxt = cur ^ 1;
#if USE_TDM
    if (wave == 0 && m0 + BT < NPIX) {   // async prefetch of next tiles
      tdm_load_2d((uint32_t)(uintptr_t)&sX[nxt][0], xt + (size_t)(m0 + BT) * CCH,
                  CCH, NPIX, CCH, CCH, BT, 6, 3);
      tdm_load_2d((uint32_t)(uintptr_t)&sVb[nxt][0], vb + (m0 + BT),
                  NPIX, CCH, NPIX, BT, CCH, 4, 3);
    }
#endif

    // ---- matmul 1: two 16x16 S tiles per wave, K=256 ----
    v8f S0 = vzero, S1 = vzero;
#pragma unroll
    for (int ks = 0; ks < 8; ++ks) {
      const v16h b0 = frag_b(sX[cur], mi0 * 16, STRX, ks * 32, r, hi);
      const v16h b1 = frag_b(sX[cur], (mi0 + 1) * 16, STRX, ks * 32, r, hi);
      S0 = wmma_f16(Amm1[ks], b0, S0);
      S1 = wmma_f16(Amm1[ks], b1, S1);
    }
    // relu^2, store transposed into sP[n][m] (C/D layout: M = e + 8*hi, N = r)
#pragma unroll
    for (int e = 0; e < 8; ++e) {
      const int nl = ni * 16 + e + 8 * hi;
      float s0 = fmaxf(S0[e], 0.f); s0 *= s0;
      float s1 = fmaxf(S1[e], 0.f); s1 *= s1;
      sP[nl * STRV + mi0 * 16 + r]       = (half_t)s0;
      sP[nl * STRV + (mi0 + 1) * 16 + r] = (half_t)s1;
    }
    __syncthreads();

    // ---- matmul 2: O[c0..c0+31, 0..63] += V * P^T, K = 64 ----
#pragma unroll
    for (int ks = 0; ks < 2; ++ks) {
      const v16h A0 = frag_a(sVb[cur], c0, STRV, ks * 32, r, hi);
      const v16h A1 = frag_a(sVb[cur], c0 + 16, STRV, ks * 32, r, hi);
      v16h Bf[4];
#pragma unroll
      for (int nj = 0; nj < 4; ++nj) Bf[nj] = frag_b(sP, nj * 16, STRV, ks * 32, r, hi);
#pragma unroll
      for (int nj = 0; nj < 4; ++nj) {
        Oacc[0][nj] = wmma_f16(A0, Bf[nj], Oacc[0][nj]);
        Oacc[1][nj] = wmma_f16(A1, Bf[nj], Oacc[1][nj]);
      }
    }

#if USE_TDM
    if (wave == 0) __builtin_amdgcn_s_wait_tensorcnt(0);  // next tiles landed
    __syncthreads();
#else
    __syncthreads();
    if (m0 + BT < NPIX) {
      for (int idx = tid; idx < (CCH * BT) / 8; idx += 256) {
        const int row = idx >> 5, cv = (idx & 31) * 8;
        *(uint4*)&sX[nxt][row * STRX + cv] =
            *(const uint4*)&xt[(size_t)(m0 + BT + row) * CCH + cv];
        const int c = idx >> 3, mv = (idx & 7) * 8;
        *(uint4*)&sVb[nxt][c * STRV + mv] =
            *(const uint4*)&vb[(size_t)c * NPIX + m0 + BT + mv];
      }
    }
    __syncthreads();
#endif
    cur = nxt;
  }

  // Writeback: element (M = e + 8*hi, N = r) of each 16x16 tile.
  float* ob = out + (size_t)b * CCH * NPIX;
#pragma unroll
  for (int ci = 0; ci < 2; ++ci)
#pragma unroll
    for (int nj = 0; nj < 4; ++nj)
#pragma unroll
      for (int e = 0; e < 8; ++e) {
        const int c = c0 + ci * 16 + e + 8 * hi;
        const int n = n0 + nj * 16 + r;
        ob[(size_t)c * NPIX + n] = Oacc[ci][nj][e];
      }
}

// ---------------- launch ----------------

extern "C" void kernel_launch(void* const* d_in, const int* in_sizes, int n_in,
                              void* d_out, int out_size, void* d_ws, size_t ws_size,
                              hipStream_t stream) {
  (void)in_sizes; (void)n_in; (void)out_size; (void)ws_size;
  const float* x    = (const float*)d_in[0];   // [4,256,64,64]
  const float* W    = (const float*)d_in[1];   // [256,256]
  const float* bias = (const float*)d_in[2];   // [256]
  float* out = (float*)d_out;                  // [4,256,64,64]

  half_t* xnT = (half_t*)d_ws;                      // 8 MB, pixel-major
  half_t* val = xnT + (size_t)4 * CCH * NPIX;       // 8 MB, channel-major

  norm_x_kernel<<<(4 * NPIX) / 256, 256, 0, stream>>>(x, xnT);
  dim3 grid(NPIX / BT, 4);
  conv_value_kernel<<<grid, 256, 0, stream>>>(x, W, bias, val);
  prop_attn_kernel<<<grid, 256, 0, stream>>>(xnT, val, out);
}
